// MoE_31542239821971
// MI455X (gfx1250) — matrix operand order repaired and checked by reading, not compiled, not run on hardware
//
#include <hip/hip_runtime.h>
#include <hip/hip_bf16.h>

// ---------------------------------------------------------------------------
// MoE (E=8, D=1024, H=4096, K=2, N=8192) — routed top-2 bf16-WMMA pipeline.
// 32-row M-tiles: each packed B tile feeds 2 WMMAs -> half the L2 weight
// traffic of the 16-row version. LDS ~83KB/WG (legal: CDNA5 allows 320KB).
// ---------------------------------------------------------------------------

#define N_TOK 8192
#define D_DIM 1024
#define H_DIM 4096
#define E_NUM 8
#define TILE_M 32                 // rows (token-expert pairs) per workgroup
#define HC    256                 // H chunk held in LDS per iteration
#define CT1   (H_DIM / 16)        // 256 column tiles of W1
#define K32_1 (D_DIM / 32)        // 32 k-tiles of W1
#define CT2   (D_DIM / 16)        // 64 column tiles of W2
#define K32_2 (H_DIM / 32)        // 128 k-tiles of W2
#define XS_STRIDE (D_DIM + 8)     // halves; 2064B rows -> conflict-free 16B reads
#define HS_STRIDE (HC + 8)        // halves; 528B rows  -> conflict-free 16B reads
#define ROWS_PER_EXP 8192
#define TILES_PER_EXP (ROWS_PER_EXP / TILE_M)   // 256

typedef __attribute__((ext_vector_type(16))) __bf16 v16bf;
typedef __attribute__((ext_vector_type(8)))  __bf16 v8bf;
typedef __attribute__((ext_vector_type(8)))  float  v8f;

struct bf16x4 { __bf16 a, b, c, d; };

// ---------------------------------------------------------------------------
// 0) zero output + routing cursors
// ---------------------------------------------------------------------------
__global__ __launch_bounds__(256) void zero_kernel(float* __restrict__ out,
                                                   int* __restrict__ cursor,
                                                   int n4) {
  int i = blockIdx.x * 256 + threadIdx.x;
  if (i < n4) {
    float4 z; z.x = 0.f; z.y = 0.f; z.z = 0.f; z.w = 0.f;
    reinterpret_cast<float4*>(out)[i] = z;
  }
  if (blockIdx.x == 0 && threadIdx.x < E_NUM) cursor[threadIdx.x] = 0;
}

// ---------------------------------------------------------------------------
// 1) x (fp32, row major) -> bf16 row major
// ---------------------------------------------------------------------------
__global__ __launch_bounds__(256) void convx_kernel(const float* __restrict__ x,
                                                    __bf16* __restrict__ xbf,
                                                    int n4) {
  int i = blockIdx.x * 256 + threadIdx.x;
  if (i < n4) {
    float4 v = reinterpret_cast<const float4*>(x)[i];
    bf16x4 o;
    o.a = (__bf16)v.x; o.b = (__bf16)v.y; o.c = (__bf16)v.z; o.d = (__bf16)v.w;
    reinterpret_cast<bf16x4*>(xbf)[i] = o;
  }
}

// ---------------------------------------------------------------------------
// 2) Pack fp32 weight [E][KROWS][NCOLS] into bf16 WMMA B-tiles:
//    tile (kt, ct) is 32x16, stored as 32 lanes x 16 halves contiguous (1KB),
//    lane l: col = ct*16 + (l&15), half t <-> K = kt*32 + (l>>4)*8 + (t&7) + (t>>3)*16.
//    Tile order: [e][ct][kt] so a wave's K-loop streams contiguous memory.
// ---------------------------------------------------------------------------
template <int KROWS, int NCOLS>
__global__ __launch_bounds__(256) void packw_kernel(const float* __restrict__ W,
                                                    __bf16* __restrict__ Wp) {
  constexpr int CT  = NCOLS / 16;
  constexpr int K32 = KROWS / 32;
  __shared__ float tile[32][17];
  const int e   = blockIdx.x / CT;
  const int ct  = blockIdx.x % CT;
  const int tid = threadIdx.x;
  const int kkR = tid >> 3;            // source row within tile (0..31)
  const int ccR = (tid & 7) * 2;       // source col pair (0,2,..,14)
  const int lW  = tid >> 3;            // dest lane (0..31)
  const int tW  = (tid * 2) & 15;      // dest half index (even)
  for (int kt = 0; kt < K32; ++kt) {
    const float* src =
        W + ((size_t)e * KROWS + (size_t)kt * 32 + kkR) * NCOLS + ct * 16 + ccR;
    tile[kkR][ccR]     = src[0];
    tile[kkR][ccR + 1] = src[1];
    __syncthreads();
    const int kk0 = ((lW >> 4) << 3) + (tW & 7) + ((tW >> 3) << 4);
    float v0 = tile[kk0][lW & 15];
    float v1 = tile[kk0 + 1][lW & 15];
    unsigned short u0 = __builtin_bit_cast(unsigned short, (__bf16)v0);
    unsigned short u1 = __builtin_bit_cast(unsigned short, (__bf16)v1);
    unsigned int pk = (unsigned int)u0 | ((unsigned int)u1 << 16);
    reinterpret_cast<unsigned int*>(
        Wp + (((size_t)blockIdx.x) * K32 + kt) * 512)[tid] = pk;
    __syncthreads();
  }
}

// ---------------------------------------------------------------------------
// 3) Gating: one wave32 per token. logits = x@Wg + bg, softmax, top-2
//    (weights NOT renormalized; strict > keeps lowest index on ties).
// ---------------------------------------------------------------------------
__global__ __launch_bounds__(256) void gate_kernel(const float* __restrict__ x,
                                                   const float* __restrict__ Wg,
                                                   const float* __restrict__ bg,
                                                   int* __restrict__ topIdx,
                                                   float* __restrict__ topW) {
  const int wave = threadIdx.x >> 5;
  const int lane = threadIdx.x & 31;
  const int n = blockIdx.x * 8 + wave;
  float acc[E_NUM] = {0.f, 0.f, 0.f, 0.f, 0.f, 0.f, 0.f, 0.f};
  for (int d0 = 0; d0 < D_DIM; d0 += 32) {
    float xv = x[(size_t)n * D_DIM + d0 + lane];
    const float* wg = Wg + (size_t)(d0 + lane) * E_NUM;
#pragma unroll
    for (int e = 0; e < E_NUM; ++e) acc[e] += xv * wg[e];
  }
#pragma unroll
  for (int off = 16; off > 0; off >>= 1) {
#pragma unroll
    for (int e = 0; e < E_NUM; ++e) acc[e] += __shfl_xor(acc[e], off, 32);
  }
  if (lane == 0) {
    float m = -3.0e38f;
#pragma unroll
    for (int e = 0; e < E_NUM; ++e) { acc[e] += bg[e]; m = fmaxf(m, acc[e]); }
    float w[E_NUM];
    float s = 0.f;
#pragma unroll
    for (int e = 0; e < E_NUM; ++e) { w[e] = expf(acc[e] - m); s += w[e]; }
    float inv = 1.0f / s;
#pragma unroll
    for (int e = 0; e < E_NUM; ++e) w[e] *= inv;
    int i0 = 0; float w0 = w[0];
#pragma unroll
    for (int e = 1; e < E_NUM; ++e) { if (w[e] > w0) { w0 = w[e]; i0 = e; } }
    int i1 = (i0 == 0) ? 1 : 0; float w1 = w[i1];
#pragma unroll
    for (int e = 0; e < E_NUM; ++e) {
      if (e != i0 && w[e] > w1) { w1 = w[e]; i1 = e; }
    }
    topIdx[n * 2 + 0] = i0; topIdx[n * 2 + 1] = i1;
    topW[n * 2 + 0] = w0;   topW[n * 2 + 1] = w1;
  }
}

// ---------------------------------------------------------------------------
// 4) Scatter token -> per-expert row lists (fixed 8192-row segments)
// ---------------------------------------------------------------------------
__global__ __launch_bounds__(256) void scatter_kernel(
    const int* __restrict__ topIdx, const float* __restrict__ topW,
    int* __restrict__ cursor, int* __restrict__ rows, float* __restrict__ rowW) {
  int n = blockIdx.x * 256 + threadIdx.x;
  if (n < N_TOK) {
#pragma unroll
    for (int k = 0; k < 2; ++k) {
      int e = topIdx[n * 2 + k];
      int pos = atomicAdd(&cursor[e], 1);
      rows[e * ROWS_PER_EXP + pos] = n;
      rowW[e * ROWS_PER_EXP + pos] = topW[n * 2 + k];
    }
  }
}

// ---------------------------------------------------------------------------
// 5) Fused routed FFN: one WG (8 wave32) per 32-row tile of one expert.
//    Two 16-row groups share every B tile (2 WMMAs per global B fetch).
//    h = relu(X@W1+b1) per 256-wide H chunk in LDS, then out += h@W2 partials
//    kept in registers; epilogue: out[tok] += w * (acc + b2) via atomic f32.
// ---------------------------------------------------------------------------
__global__ __launch_bounds__(256) void ffn_kernel(
    const __bf16* __restrict__ xbf, const __bf16* __restrict__ W1p,
    const __bf16* __restrict__ W2p, const float* __restrict__ b1,
    const float* __restrict__ b2, const int* __restrict__ rows,
    const float* __restrict__ rowW, const int* __restrict__ cursor,
    float* __restrict__ out) {
  __shared__ __align__(16) __bf16 Xs[TILE_M * XS_STRIDE];   // 66,048 B
  __shared__ __align__(16) __bf16 Hs[TILE_M * HS_STRIDE];   // 16,896 B
  __shared__ int   rowsS[TILE_M];
  __shared__ float wS[TILE_M];

  const int e   = blockIdx.x >> 8;       // 256 tiles per expert
  const int ti  = blockIdx.x & 255;
  const int cnt = cursor[e];
  const int base = ti * TILE_M;
  if (base >= cnt) return;               // uniform across block

  const int tid = threadIdx.x;
  if (tid < TILE_M) {
    int r = base + tid;
    bool v = r < cnt;
    rowsS[tid] = v ? rows[e * ROWS_PER_EXP + r] : -1;
    wS[tid]    = v ? rowW[e * ROWS_PER_EXP + r] : 0.0f;
  }
  __syncthreads();

  // ---- gather 32 token rows of bf16 x into LDS (zeros for padding rows)
  {
    const int r  = tid >> 3;     // row 0..31 (8 threads per row)
    const int t8 = tid & 7;
    const int tok = rowsS[r];
    const __bf16* src = xbf + (size_t)(tok < 0 ? 0 : tok) * D_DIM;
#pragma unroll
    for (int j = 0; j < 16; ++j) {
      const int off = j * 64 + t8 * 8;   // halves, 16B chunks
      uint4 val; val.x = 0u; val.y = 0u; val.z = 0u; val.w = 0u;
      if (tok >= 0) val = *reinterpret_cast<const uint4*>(src + off);
      *reinterpret_cast<uint4*>(&Xs[r * XS_STRIDE + off]) = val;
    }
  }

  const int wave   = tid >> 5;
  const int lane   = tid & 31;
  const int lane16 = lane & 15;
  const int laneHi = lane >> 4;          // 0 or 1

  v8f accD[16];                          // [p][g] -> p*2+g
#pragma unroll
  for (int p = 0; p < 16; ++p) accD[p] = {};

  __syncthreads();

  for (int ch = 0; ch < H_DIM / HC; ++ch) {
    const int hBase = ch * HC;
    // ---- FFN1: each wave computes 2 column tiles (16 cols) x 2 row groups
#pragma unroll
    for (int q = 0; q < 2; ++q) {
      const int cLocal = (wave * 2 + q) * 16;        // 0..255
      const int cGlob  = hBase + cLocal;
      const int ct     = cGlob >> 4;
      v8f acc0 = {};
      v8f acc1 = {};
      const __bf16* btile =
          W1p + (((size_t)e * CT1 + ct) * K32_1) * 512 + lane * 16;
      for (int kt = 0; kt < K32_1; ++kt) {
        const int aOff = kt * 32 + laneHi * 8;
        v16bf b = *reinterpret_cast<const v16bf*>(btile + (size_t)kt * 512);
        v8bf a0l = *reinterpret_cast<const v8bf*>(&Xs[lane16 * XS_STRIDE + aOff]);
        v8bf a0h = *reinterpret_cast<const v8bf*>(&Xs[lane16 * XS_STRIDE + aOff + 16]);
        v16bf a0 = __builtin_shufflevector(a0l, a0h, 0, 1, 2, 3, 4, 5, 6, 7, 8,
                                           9, 10, 11, 12, 13, 14, 15);
        acc0 = __builtin_amdgcn_wmma_f32_16x16x32_bf16(false, a0, false, b,
                                                       (short)0, acc0, false, false);
        v8bf a1l = *reinterpret_cast<const v8bf*>(
            &Xs[(16 + lane16) * XS_STRIDE + aOff]);
        v8bf a1h = *reinterpret_cast<const v8bf*>(
            &Xs[(16 + lane16) * XS_STRIDE + aOff + 16]);
        v16bf a1 = __builtin_shufflevector(a1l, a1h, 0, 1, 2, 3, 4, 5, 6, 7, 8,
                                           9, 10, 11, 12, 13, 14, 15);
        acc1 = __builtin_amdgcn_wmma_f32_16x16x32_bf16(false, a1, false, b,
                                                       (short)0, acc1, false, false);
      }
      const float bias = b1[e * H_DIM + cGlob + lane16];
#pragma unroll
      for (int v = 0; v < 8; ++v) {
        const int M = v + laneHi * 8;
        float h0 = acc0[v] + bias;
        h0 = h0 > 0.f ? h0 : 0.f;
        Hs[M * HS_STRIDE + cLocal + lane16] = (__bf16)h0;
        float h1 = acc1[v] + bias;
        h1 = h1 > 0.f ? h1 : 0.f;
        Hs[(16 + M) * HS_STRIDE + cLocal + lane16] = (__bf16)h1;
      }
    }
    __syncthreads();
    // ---- FFN2 partial: each wave owns 128 output cols (8 tiles) x 2 groups
#pragma unroll
    for (int p = 0; p < 8; ++p) {
      const int colD = wave * 128 + p * 16;
      const int ct2  = colD >> 4;
      const __bf16* btile =
          W2p + (((size_t)e * CT2 + ct2) * K32_2 + (hBase >> 5)) * 512 + lane * 16;
      v8f acc0 = accD[p * 2 + 0];
      v8f acc1 = accD[p * 2 + 1];
      for (int kt = 0; kt < HC / 32; ++kt) {
        const int aOff = kt * 32 + laneHi * 8;
        v16bf b = *reinterpret_cast<const v16bf*>(btile + (size_t)kt * 512);
        v8bf a0l = *reinterpret_cast<const v8bf*>(&Hs[lane16 * HS_STRIDE + aOff]);
        v8bf a0h = *reinterpret_cast<const v8bf*>(&Hs[lane16 * HS_STRIDE + aOff + 16]);
        v16bf a0 = __builtin_shufflevector(a0l, a0h, 0, 1, 2, 3, 4, 5, 6, 7, 8,
                                           9, 10, 11, 12, 13, 14, 15);
        acc0 = __builtin_amdgcn_wmma_f32_16x16x32_bf16(false, a0, false, b,
                                                       (short)0, acc0, false, false);
        v8bf a1l = *reinterpret_cast<const v8bf*>(
            &Hs[(16 + lane16) * HS_STRIDE + aOff]);
        v8bf a1h = *reinterpret_cast<const v8bf*>(
            &Hs[(16 + lane16) * HS_STRIDE + aOff + 16]);
        v16bf a1 = __builtin_shufflevector(a1l, a1h, 0, 1, 2, 3, 4, 5, 6, 7, 8,
                                           9, 10, 11, 12, 13, 14, 15);
        acc1 = __builtin_amdgcn_wmma_f32_16x16x32_bf16(false, a1, false, b,
                                                       (short)0, acc1, false, false);
      }
      accD[p * 2 + 0] = acc0;
      accD[p * 2 + 1] = acc1;
    }
    __syncthreads();  // Hs will be overwritten next chunk
  }

  // ---- epilogue: out[tok] += w * (acc + b2)   (exactly 2 adds/elem -> det.)
#pragma unroll
  for (int p = 0; p < 8; ++p) {
    const int col = wave * 128 + p * 16 + lane16;
    const float bias2 = b2[e * D_DIM + col];
#pragma unroll
    for (int g = 0; g < 2; ++g) {
#pragma unroll
      for (int v = 0; v < 8; ++v) {
        const int M = g * 16 + v + laneHi * 8;
        const int tok = rowsS[M];
        if (tok >= 0) {
          atomicAdd(&out[(size_t)tok * D_DIM + col],
                    wS[M] * (accD[p * 2 + g][v] + bias2));
        }
      }
    }
  }
}

// ---------------------------------------------------------------------------
// Launch
// ---------------------------------------------------------------------------
extern "C" void kernel_launch(void* const* d_in, const int* in_sizes, int n_in,
                              void* d_out, int out_size, void* d_ws, size_t ws_size,
                              hipStream_t stream) {
  const float* x  = (const float*)d_in[0];
  const float* Wg = (const float*)d_in[1];
  const float* bg = (const float*)d_in[2];
  const float* W1 = (const float*)d_in[3];
  const float* b1 = (const float*)d_in[4];
  const float* W2 = (const float*)d_in[5];
  const float* b2 = (const float*)d_in[6];
  float* out = (float*)d_out;

  // workspace layout (bytes)
  char* ws = (char*)d_ws;
  const size_t OFF_XBF  = 0;                                   // 16 MB
  const size_t OFF_W1P  = OFF_XBF + (size_t)N_TOK * D_DIM * 2; // 64 MB
  const size_t OFF_W2P  = OFF_W1P + (size_t)E_NUM * D_DIM * H_DIM * 2;
  const size_t OFF_TOPI = OFF_W2P + (size_t)E_NUM * H_DIM * D_DIM * 2;
  const size_t OFF_TOPW = OFF_TOPI + (size_t)N_TOK * 2 * 4;
  const size_t OFF_CUR  = OFF_TOPW + (size_t)N_TOK * 2 * 4;
  const size_t OFF_ROWS = OFF_CUR + 256;
  const size_t OFF_ROWW = OFF_ROWS + (size_t)E_NUM * ROWS_PER_EXP * 4;

  __bf16* xbf  = (__bf16*)(ws + OFF_XBF);
  __bf16* W1p  = (__bf16*)(ws + OFF_W1P);
  __bf16* W2p  = (__bf16*)(ws + OFF_W2P);
  int*   topIdx = (int*)(ws + OFF_TOPI);
  float* topW   = (float*)(ws + OFF_TOPW);
  int*   cursor = (int*)(ws + OFF_CUR);
  int*   rows   = (int*)(ws + OFF_ROWS);
  float* rowW   = (float*)(ws + OFF_ROWW);

  const int n4 = (N_TOK * D_DIM) / 4;  // 2,097,152

  zero_kernel<<<n4 / 256, 256, 0, stream>>>(out, cursor, n4);
  convx_kernel<<<n4 / 256, 256, 0, stream>>>(x, xbf, n4);
  packw_kernel<D_DIM, H_DIM><<<E_NUM * CT1, 256, 0, stream>>>(W1, W1p);
  packw_kernel<H_DIM, D_DIM><<<E_NUM * CT2, 256, 0, stream>>>(W2, W2p);
  gate_kernel<<<N_TOK / 8, 256, 0, stream>>>(x, Wg, bg, topIdx, topW);
  scatter_kernel<<<N_TOK / 256, 256, 0, stream>>>(topIdx, topW, cursor, rows, rowW);
  ffn_kernel<<<E_NUM * TILES_PER_EXP, 256, 0, stream>>>(xbf, W1p, W2p, b1, b2,
                                                        rows, rowW, cursor, out);
}
